// BiLSTMEncoder_48026324303986
// MI455X (gfx1250) — matrix operand order repaired and checked by reading, not compiled, not run on hardware
//
#include <hip/hip_runtime.h>

// ---------------------------------------------------------------------------
// CDNA5 (gfx1250) implementation of the BiLSTM/Transformer/GATv2 pipeline.
// Dense GEMMs use v_wmma_f32_16x16x32_bf16 (bf16 in, fp32 accumulate).
// Fragments live in WMMA-native layout so each operand is one contiguous
// 32B load per lane; GEMM K-loop is double-buffered (1 barrier / K-step).
// ---------------------------------------------------------------------------

typedef __attribute__((ext_vector_type(16))) __bf16 v16bf;
typedef __attribute__((ext_vector_type(4)))  __bf16 v4bf;
typedef __attribute__((ext_vector_type(8)))  float  v8f;

__device__ inline __bf16 f2bf(float f) {
  unsigned u = __float_as_uint(f);
  u += 0x7fffu + ((u >> 16) & 1u);          // round-to-nearest-even
  unsigned short s = (unsigned short)(u >> 16);
  return __builtin_bit_cast(__bf16, s);
}
__device__ inline __bf16 bf_zero() {
  unsigned short s = 0;
  return __builtin_bit_cast(__bf16, s);
}

// WMMA 16x16x32 bf16 fragment geometry (ISA 7.12.2):
//  A (16x32 MxK): lane = half*16 + (m&15); elem e -> k_off = (e&7)+8*half+16*(e>=8)
//     inverse: half=(rem>>3)&1, e=(rem&7)+8*(rem>=16)
//  B (32x16 KxN): lane = half*16 + (n&15); elem e -> k_off = (e&7)+8*half+16*(e>=8)
//     inverse: half = (rem>>3)&1, e = (rem&7)+8*(rem>=16)
//  C/D: row = j + 8*(lane>>4), col = lane&15

// ---------------------------------------------------------------------------
// Generic WMMA GEMM: C[M,N] = act(A[M,K] @ B[K,N] + bias) (+ res)
// act: 0 none, 1 relu, 2 leaky(0.01), 3 gelu(exact)
// 256 threads = 8 waves; block tile 64x64; each wave: 2 M-tiles x 1 N-tile.
// Double-buffered LDS staging; interior K-steps use unconditional float4.
// ---------------------------------------------------------------------------
__global__ __launch_bounds__(256)
void wmma_gemm_kernel(const float* __restrict__ A, const float* __restrict__ B,
                      const float* __restrict__ bias, const float* __restrict__ res,
                      float* __restrict__ C, int M, int N, int K, int act) {
  __shared__ __bf16 Af[2][4 * 32 * 16];   // [buf][mt][lane][e]
  __shared__ __bf16 Bf[2][4 * 32 * 16];   // [buf][nt][lane][e]
  int tid  = threadIdx.x;
  int lane = tid & 31, wave = tid >> 5;
  int wn = wave & 3, wg = wave >> 2;           // N-tile, M-tile-pair
  int rowBase = blockIdx.y * 64;
  int colBase = blockIdx.x * 64;
  bool rcInterior = (rowBase + 64 <= M) && (colBase + 64 <= N);

  // per-thread staging registers + fixed tile coordinates
  float4 ra[2], rb[2];
  int rA[2], c4A[2], rB[2], c4B[2];
#pragma unroll
  for (int i = 0; i < 2; ++i) {
    int vv = tid + i * 256;
    rA[i] = vv >> 3;  c4A[i] = (vv & 7) << 2;    // A: 64 rows x 32 k
    rB[i] = vv >> 4;  c4B[i] = (vv & 15) << 2;   // B: 32 k x 64 n
  }

  auto fetch = [&](int k0) {
    if (rcInterior && (k0 + 32 <= K)) {
#pragma unroll
      for (int i = 0; i < 2; ++i)
        ra[i] = *(const float4*)(A + (long long)(rowBase + rA[i]) * K + k0 + c4A[i]);
#pragma unroll
      for (int i = 0; i < 2; ++i)
        rb[i] = *(const float4*)(B + (long long)(k0 + rB[i]) * N + colBase + c4B[i]);
    } else {
#pragma unroll
      for (int i = 0; i < 2; ++i) {
        float t0 = 0.f, t1 = 0.f, t2 = 0.f, t3 = 0.f;
        int gr = rowBase + rA[i];
        if (gr < M) {
          int gk = k0 + c4A[i];
          const float* ap = A + (long long)gr * K;
          if (gk + 0 < K) t0 = ap[gk + 0];
          if (gk + 1 < K) t1 = ap[gk + 1];
          if (gk + 2 < K) t2 = ap[gk + 2];
          if (gk + 3 < K) t3 = ap[gk + 3];
        }
        ra[i] = make_float4(t0, t1, t2, t3);
      }
#pragma unroll
      for (int i = 0; i < 2; ++i) {
        float t0 = 0.f, t1 = 0.f, t2 = 0.f, t3 = 0.f;
        int gk = k0 + rB[i];
        if (gk < K) {
          int gc = colBase + c4B[i];
          const float* bp = B + (long long)gk * N;
          if (gc + 0 < N) t0 = bp[gc + 0];
          if (gc + 1 < N) t1 = bp[gc + 1];
          if (gc + 2 < N) t2 = bp[gc + 2];
          if (gc + 3 < N) t3 = bp[gc + 3];
        }
        rb[i] = make_float4(t0, t1, t2, t3);
      }
    }
  };

  auto commit = [&](int buf) {
#pragma unroll
    for (int i = 0; i < 2; ++i) {
      int r = rA[i], c4 = c4A[i];
      int mt = r >> 4, lm = r & 15;
      int half = (c4 >> 3) & 1;
      int e = (c4 & 7) + ((c4 >> 4) << 3);       // 4 consecutive e, 4-aligned
      v4bf pk;
      pk[0] = f2bf(ra[i].x); pk[1] = f2bf(ra[i].y);
      pk[2] = f2bf(ra[i].z); pk[3] = f2bf(ra[i].w);
      *(v4bf*)&Af[buf][(mt * 32 + half * 16 + lm) * 16 + e] = pk;
    }
#pragma unroll
    for (int i = 0; i < 2; ++i) {
      int r = rB[i], c4 = c4B[i];
      int half = (r >> 3) & 1, e = (r & 7) + ((r >> 4) << 3);
      int nt = c4 >> 4, ln = c4 & 15;
      __bf16* dp = &Bf[buf][(nt * 32 + half * 16 + ln) * 16 + e];
      dp[0]  = f2bf(rb[i].x);
      dp[16] = f2bf(rb[i].y);
      dp[32] = f2bf(rb[i].z);
      dp[48] = f2bf(rb[i].w);
    }
  };

  v8f acc0, acc1;
  {
    int c = colBase + wn * 16 + (lane & 15);
    float bv = (bias && c < N) ? bias[c] : 0.f;
#pragma unroll
    for (int j = 0; j < 8; ++j) { acc0[j] = bv; acc1[j] = bv; }
  }

  fetch(0);
  commit(0);
  __syncthreads();
  int cur = 0;
  for (int k0 = 0; k0 < K; k0 += 32) {
    bool hasNext = (k0 + 32 < K);
    if (hasNext) fetch(k0 + 32);               // overlap next fetch with math
    v16bf a0 = *(const v16bf*)&Af[cur][((wg * 2 + 0) * 32 + lane) * 16];
    v16bf a1 = *(const v16bf*)&Af[cur][((wg * 2 + 1) * 32 + lane) * 16];
    v16bf bb = *(const v16bf*)&Bf[cur][(wn * 32 + lane) * 16];
    acc0 = __builtin_amdgcn_wmma_f32_16x16x32_bf16(false, a0, false, bb, (short)0, acc0, false, false);
    acc1 = __builtin_amdgcn_wmma_f32_16x16x32_bf16(false, a1, false, bb, (short)0, acc1, false, false);
    if (hasNext) commit(cur ^ 1);              // safe: cur^1 last read 1 iter ago
    __syncthreads();
    cur ^= 1;
  }

  int half = lane >> 4;
  int cn = colBase + wn * 16 + (lane & 15);
  if (cn < N) {
#pragma unroll
    for (int pairi = 0; pairi < 2; ++pairi) {
      int tileRow = rowBase + (wg * 2 + pairi) * 16;
#pragma unroll
      for (int j = 0; j < 8; ++j) {
        int r = tileRow + j + 8 * half;
        if (r < M) {
          float v = (pairi == 0) ? acc0[j] : acc1[j];
          if (act == 1)      v = fmaxf(v, 0.f);
          else if (act == 2) v = (v > 0.f) ? v : 0.01f * v;
          else if (act == 3) v = 0.5f * v * (1.f + erff(v * 0.70710678118f));
          long long o = (long long)r * N + cn;
          if (res) v += res[o];
          C[o] = v;
        }
      }
    }
  }
}

// ---------------------------------------------------------------------------
// Repack whh [500,2000] fp32 -> bf16 fragment-layout buffer, zero-padded:
//   packed[ nt(32) ][ k0(16) ][ gate(4) ][ lane(32) ][ e(16) ]
// (k = k0*32 + 8*(lane>>4) + (e&7) + 16*(e>>3) ; col-within-gate = nt*16 + (lane&15))
// ---------------------------------------------------------------------------
__global__ void lstm_pack_whh(const float* __restrict__ whh, __bf16* __restrict__ outp) {
  int i = blockIdx.x * 256 + threadIdx.x;
  if (i >= (1 << 20)) return;
  int e    = i & 15;
  int lane = (i >> 4) & 31;
  int g    = (i >> 9) & 3;
  int k0   = (i >> 11) & 15;
  int nt   = i >> 15;
  int k    = k0 * 32 + 8 * (lane >> 4) + (e & 7) + 16 * (e >> 3);
  int colg = nt * 16 + (lane & 15);
  float v = (k < 500 && colg < 500) ? whh[(long long)k * 2000 + g * 500 + colg] : 0.f;
  outp[i] = f2bf(v);
}

// ---------------------------------------------------------------------------
// Persistent single-workgroup LSTM direction kernel.
// h kept in LDS in A-fragment layout [mt(3)][k0(16)][lane(32)][e(16)] bf16,
// double-buffered; c fp32 in LDS. Per step: g = pre[t] + h @ whh (WMMA),
// gates fused. 512 threads = 16 waves; 96 (mt,nt) pairs, 6 per wave.
// dynamic LDS: 2*24576 bf16 + 40*512 fp32 = 180224 B (WGP has 320KB).
// ---------------------------------------------------------------------------
__global__ __launch_bounds__(512)
void lstm_dir_kernel(const float* __restrict__ pre,      // [T,40,2000]
                     const __bf16* __restrict__ whhp,    // packed (see above)
                     float* __restrict__ out,            // rows (t*40+b), ld=ldo
                     int T, int reverse, int ldo, int coff) {
  extern __shared__ char smem[];
  __bf16* hbase = (__bf16*)smem;                          // [2][3*16*32*16]
  float*  cs    = (float*)(smem + 2 * 24576 * 2);         // [40][512]
  const int HB = 24576;

  int tid = threadIdx.x, lane = tid & 31, wave = tid >> 5;
  for (int i = tid; i < 2 * HB; i += 512) hbase[i] = bf_zero();
  for (int i = tid; i < 40 * 512; i += 512) cs[i] = 0.f;
  __syncthreads();

  int cur = 0;
  for (int step = 0; step < T; ++step) {
    int t = reverse ? (T - 1 - step) : step;
    const float* pre_t = pre + (long long)t * 40 * 2000;
    const __bf16* hcur = hbase + cur * HB;
    __bf16* hnxt = hbase + (cur ^ 1) * HB;

    for (int p = wave; p < 96; p += 16) {
      int mt = p >> 5;              // 0..2
      int nt = p & 31;              // 0..31
      int n16 = lane & 15, half = lane >> 4;
      int col = nt * 16 + n16;      // column within one gate (valid < 500)

      v8f acc[4];
#pragma unroll
      for (int g = 0; g < 4; ++g)
#pragma unroll
        for (int j = 0; j < 8; ++j) acc[g][j] = 0.f;

      for (int k0 = 0; k0 < 16; ++k0) {
        v16bf af = *(const v16bf*)&hcur[((mt * 16 + k0) * 32 + lane) * 16];
        const __bf16* wp = whhp + (((long long)(nt * 16 + k0) * 4) * 32 + lane) * 16;
        // prefetch next k0's packed-weight block (global_prefetch_b8)
        __builtin_prefetch((const void*)(wp + 4 * 32 * 16), 0, 1);
#pragma unroll
        for (int g = 0; g < 4; ++g) {
          v16bf bfr = *(const v16bf*)(wp + (long long)g * 32 * 16);
          acc[g] = __builtin_amdgcn_wmma_f32_16x16x32_bf16(false, af, false, bfr, (short)0, acc[g], false, false);
        }
      }

      if (col < 500) {
#pragma unroll
        for (int j = 0; j < 8; ++j) {
          int m = mt * 16 + j + 8 * half;
          if (m < 40) {
            const float* pr = pre_t + (long long)m * 2000;
            float gi = acc[0][j] + pr[col];
            float gf = acc[1][j] + pr[col + 500];
            float gg = acc[2][j] + pr[col + 1000];
            float go = acc[3][j] + pr[col + 1500];
            float si = 1.f / (1.f + __expf(-gi));
            float sf = 1.f / (1.f + __expf(-gf));
            float so = 1.f / (1.f + __expf(-go));
            float cv = sf * cs[m * 512 + col] + si * tanhf(gg);
            float hv = so * tanhf(cv);
            cs[m * 512 + col] = cv;
            // scatter h into next-step A-fragment slot
            int k0w = col >> 5, rem = col & 31;
            int hw = (rem >> 3) & 1;
            int ew = (rem & 7) + ((rem >> 4) << 3);
            int lw = hw * 16 + (m & 15);
            hnxt[((mt * 16 + k0w) * 32 + lw) * 16 + ew] = f2bf(hv);
            out[((long long)t * 40 + m) * ldo + coff + col] = hv;
          }
        }
      }
    }
    __syncthreads();
    cur ^= 1;
  }
}

// ---------------------------------------------------------------------------
// Conv1d k=3, pad=1, layout [B,C,L].  act: 0 none, 1 leaky(0.01)
// ---------------------------------------------------------------------------
__global__ void conv1d_k3(const float* __restrict__ x, const float* __restrict__ w,
                          const float* __restrict__ b, float* __restrict__ y,
                          int B, int Cin, int Cout, int L, int act) {
  long long idx = blockIdx.x * (long long)blockDim.x + threadIdx.x;
  long long total = (long long)B * Cout * L;
  if (idx >= total) return;
  int l = (int)(idx % L);
  long long t = idx / L;
  int o = (int)(t % Cout);
  int bb = (int)(t / Cout);
  float s = b[o];
  for (int i = 0; i < Cin; ++i) {
    const float* xp = x + ((long long)bb * Cin + i) * L + l;
    const float* wp = w + ((long long)o * Cin + i) * 3;
    if (l > 0)     s += xp[-1] * wp[0];
    s += xp[0] * wp[1];
    if (l < L - 1) s += xp[1] * wp[2];
  }
  if (act) s = (s > 0.f) ? s : 0.01f * s;
  y[idx] = s;
}

// BatchNorm over (B,L) per channel, layout [B,C,L]
__global__ __launch_bounds__(256)
void bn_stats_ch(const float* __restrict__ x, float* mu, float* var, int B, int C, int L) {
  int c = blockIdx.x, tid = threadIdx.x;
  __shared__ float ssum[256], ssq[256];
  float s = 0.f, q = 0.f;
  int n = B * L;
  for (int i = tid; i < n; i += 256) {
    int b = i / L, l = i % L;
    float v = x[((long long)b * C + c) * L + l];
    s += v; q += v * v;
  }
  ssum[tid] = s; ssq[tid] = q; __syncthreads();
  for (int st = 128; st > 0; st >>= 1) {
    if (tid < st) { ssum[tid] += ssum[tid + st]; ssq[tid] += ssq[tid + st]; }
    __syncthreads();
  }
  if (tid == 0) { float m = ssum[0] / n; mu[c] = m; var[c] = ssq[0] / n - m * m; }
}
__global__ void bn_apply_ch(const float* __restrict__ x, float* __restrict__ y,
                            const float* mu, const float* var,
                            const float* g, const float* be, int B, int C, int L) {
  long long idx = blockIdx.x * (long long)blockDim.x + threadIdx.x;
  long long total = (long long)B * C * L;
  if (idx >= total) return;
  int c = (int)((idx / L) % C);
  float v = (x[idx] - mu[c]) * rsqrtf(var[c] + 1e-5f) * g[c] + be[c];
  y[idx] = (v > 0.f) ? v : 0.01f * v;   // leaky_relu after BN
}

// BatchNorm over rows per column, layout [Nn,C]; apply includes ReLU
__global__ __launch_bounds__(256)
void bn_stats_col(const float* __restrict__ x, float* mu, float* var, int Nn, int C) {
  int c = blockIdx.x, tid = threadIdx.x;
  __shared__ float ssum[256], ssq[256];
  float s = 0.f, q = 0.f;
  for (int r = tid; r < Nn; r += 256) {
    float v = x[(long long)r * C + c];
    s += v; q += v * v;
  }
  ssum[tid] = s; ssq[tid] = q; __syncthreads();
  for (int st = 128; st > 0; st >>= 1) {
    if (tid < st) { ssum[tid] += ssum[tid + st]; ssq[tid] += ssq[tid + st]; }
    __syncthreads();
  }
  if (tid == 0) { float m = ssum[0] / Nn; mu[c] = m; var[c] = ssq[0] / Nn - m * m; }
}
__global__ void bn_apply_col(float* __restrict__ x, const float* mu, const float* var,
                             const float* g, const float* be, int Nn, int C) {
  long long idx = blockIdx.x * (long long)blockDim.x + threadIdx.x;
  if (idx >= (long long)Nn * C) return;
  int c = (int)(idx % C);
  float v = (x[idx] - mu[c]) * rsqrtf(var[c] + 1e-5f) * g[c] + be[c];
  x[idx] = fmaxf(v, 0.f);
}

// LayerNorm over last dim (block per row)
__global__ __launch_bounds__(128)
void layernorm_kernel(const float* __restrict__ x, const float* g, const float* b,
                      float* __restrict__ y, int D) {
  int row = blockIdx.x, tid = threadIdx.x;
  const float* xr = x + (long long)row * D;
  __shared__ float red[128];
  float s = 0.f;
  for (int i = tid; i < D; i += 128) s += xr[i];
  red[tid] = s; __syncthreads();
  for (int st = 64; st > 0; st >>= 1) { if (tid < st) red[tid] += red[tid + st]; __syncthreads(); }
  float mu = red[0] / D;
  __syncthreads();
  float q = 0.f;
  for (int i = tid; i < D; i += 128) { float d = xr[i] - mu; q += d * d; }
  red[tid] = q; __syncthreads();
  for (int st = 64; st > 0; st >>= 1) { if (tid < st) red[tid] += red[tid + st]; __syncthreads(); }
  float inv = rsqrtf(red[0] / D + 1e-5f);
  for (int i = tid; i < D; i += 128)
    y[(long long)row * D + i] = (xr[i] - mu) * inv * g[i] + b[i];
}

// Tiny attention: block per (batch b, head h); S=256, TOK=40, 8 heads, dh=8
__global__ __launch_bounds__(256)
void attn_kernel(const float* __restrict__ qkv, float* __restrict__ out) {
  int bh = blockIdx.x;
  int b = bh >> 3, h = bh & 7;
  __shared__ float q[40][8], k[40][8], v[40][8], s[40][40];
  int tid = threadIdx.x;
  for (int i = tid; i < 320; i += 256) {
    int n = i >> 3, d = i & 7;
    long long base = (long long)(b * 40 + n) * 192;
    q[n][d] = qkv[base + h * 8 + d];
    k[n][d] = qkv[base + 64 + h * 8 + d];
    v[n][d] = qkv[base + 128 + h * 8 + d];
  }
  __syncthreads();
  const float scale = 0.3535533906f;   // 8^-0.5
  for (int i = tid; i < 1600; i += 256) {
    int n = i / 40, m = i % 40;
    float d = 0.f;
#pragma unroll
    for (int e = 0; e < 8; ++e) d += q[n][e] * k[m][e];
    s[n][m] = d * scale;
  }
  __syncthreads();
  if (tid < 40) {
    int n = tid;
    float mx = -3.0e38f;
    for (int m = 0; m < 40; ++m) mx = fmaxf(mx, s[n][m]);
    float sum = 0.f;
    for (int m = 0; m < 40; ++m) { float e = __expf(s[n][m] - mx); s[n][m] = e; sum += e; }
    float inv = 1.f / sum;
    for (int m = 0; m < 40; ++m) s[n][m] *= inv;
  }
  __syncthreads();
  for (int i = tid; i < 320; i += 256) {
    int n = i >> 3, d = i & 7;
    float o = 0.f;
    for (int m = 0; m < 40; ++m) o += s[n][m] * v[m][d];
    out[(long long)(b * 40 + n) * 64 + h * 8 + d] = o;
  }
}

// ---------------------------------------------------------------------------
// GATv2 edge kernels
// ---------------------------------------------------------------------------
__device__ inline void atomicMaxF(float* addr, float val) {
  if (val >= 0.f) atomicMax((int*)addr, __float_as_int(val));
  else            atomicMin((unsigned int*)addr, __float_as_uint(val));
}

__global__ void fill_kernel(float* p, float v, long long n) {
  long long i = blockIdx.x * (long long)blockDim.x + threadIdx.x;
  if (i < n) p[i] = v;
}

__global__ void gat_edge_score(const float* __restrict__ xl, const float* __restrict__ xr,
                               const float* __restrict__ att,
                               const int* __restrict__ src, const int* __restrict__ dst,
                               float* __restrict__ a, float* __restrict__ mbuf,
                               int E, int H, int Cc) {
  long long i = blockIdx.x * (long long)blockDim.x + threadIdx.x;
  if (i >= (long long)E * H) return;
  int e = (int)(i / H), h = (int)(i % H);
  int s = src[e], d = dst[e];
  const float* pl = xl + (long long)s * H * Cc + h * Cc;
  const float* pr = xr + (long long)d * H * Cc + h * Cc;
  const float* pa = att + h * Cc;
  float acc = 0.f;
  for (int c = 0; c < Cc; ++c) {
    float t = pl[c] + pr[c];
    t = (t > 0.f) ? t : 0.2f * t;
    acc += t * pa[c];
  }
  a[i] = acc;
  atomicMaxF(&mbuf[(long long)d * H + h], acc);
}

__global__ void gat_edge_exp(const float* __restrict__ a, const int* __restrict__ dst,
                             const float* __restrict__ mbuf,
                             float* __restrict__ w, float* __restrict__ denom, int E, int H) {
  long long i = blockIdx.x * (long long)blockDim.x + threadIdx.x;
  if (i >= (long long)E * H) return;
  int e = (int)(i / H), h = (int)(i % H);
  int d = dst[e];
  float wv = __expf(a[i] - mbuf[(long long)d * H + h]);
  w[i] = wv;
  atomicAdd(&denom[(long long)d * H + h], wv);
}

__global__ __launch_bounds__(256)
void gat_edge_accum(const float* __restrict__ xl, const float* __restrict__ w,
                    const float* __restrict__ denom,
                    const int* __restrict__ src, const int* __restrict__ dst,
                    float* __restrict__ out, int E, int H, int Cc) {
  int e = blockIdx.x;
  int s = src[e], d = dst[e];
  int HC = H * Cc;
  for (int idx = threadIdx.x; idx < HC; idx += 256) {
    int h = idx / Cc;
    float coef = w[(long long)e * H + h] / (denom[(long long)d * H + h] + 1e-16f);
    atomicAdd(&out[(long long)d * HC + idx], xl[(long long)s * HC + idx] * coef);
  }
}

__global__ void bias_act_kernel(float* __restrict__ x, const float* __restrict__ bias,
                                long long n, int C, int relu) {
  long long i = blockIdx.x * (long long)blockDim.x + threadIdx.x;
  if (i >= n) return;
  float v = x[i] + bias[(int)(i % C)];
  if (relu) v = fmaxf(v, 0.f);
  x[i] = v;
}

// pooled[g,c] = sum_t x[(g*TOK+t), c]
__global__ void pool_kernel(const float* __restrict__ x, float* __restrict__ out,
                            int G, int TOKn, int C) {
  long long i = blockIdx.x * (long long)blockDim.x + threadIdx.x;
  if (i >= (long long)G * C) return;
  int g = (int)(i / C), c = (int)(i % C);
  float s = 0.f;
  for (int t = 0; t < TOKn; ++t) s += x[((long long)g * TOKn + t) * C + c];
  out[i] = s;
}

__global__ __launch_bounds__(256)
void rownorm_kernel(float* __restrict__ x, int C) {
  int row = blockIdx.x, tid = threadIdx.x;
  float* xr = x + (long long)row * C;
  __shared__ float red[256];
  float q = 0.f;
  for (int i = tid; i < C; i += 256) { float v = xr[i]; q += v * v; }
  red[tid] = q; __syncthreads();
  for (int st = 128; st > 0; st >>= 1) { if (tid < st) red[tid] += red[tid + st]; __syncthreads(); }
  float inv = 1.f / fmaxf(sqrtf(red[0]), 1e-12f);
  for (int i = tid; i < C; i += 256) xr[i] *= inv;
}

// ---------------------------------------------------------------------------
// Host orchestration
// ---------------------------------------------------------------------------
extern "C" void kernel_launch(void* const* d_in, const int* in_sizes, int n_in,
                              void* d_out, int out_size, void* d_ws, size_t ws_size,
                              hipStream_t stream) {
  (void)n_in; (void)out_size; (void)ws_size;
  const float* x = (const float*)d_in[0];
  const int* ei = (const int*)d_in[1];
  int E = in_sizes[1] / 2;
  const int* src = ei;
  const int* dst = ei + E;

  auto P = [&](int i) { return (const float*)d_in[i]; };
  // params flattened in dict-insertion order (leaves 3..62)
  const float *l0w1 = P(3), *l0b1 = P(4), *l0w2 = P(5), *l0b2 = P(6), *l0w3 = P(7), *l0b3 = P(8);
  const float *c11w = P(9), *c11b = P(10), *c21w = P(11), *c21b = P(12), *bn11g = P(13), *bn11b = P(14);
  const float *g1wl = P(49), *g1bl = P(50), *g1wr = P(51), *g1br = P(52), *g1att = P(53), *g1bias = P(54);
  const float *g2wl = P(55), *g2bl = P(56), *g2wr = P(57), *g2br = P(58), *g2att = P(59), *g2bias = P(60);
  const float *bn2g = P(61), *bn2b = P(62);

  // workspace layout (floats)
  float* ws = (float*)d_ws;
  float* W0 = ws;                  // 20,480,000  (pre_f / GAT scratch)
  float* W1 = W0 + 20480000;       // 20,480,000  (pre_b / GAT edge scratch)
  float* Z  = W1 + 20480000;       // 10,240,000  (h / z / nodes / g2)
  float* T0 = Z  + 10240000;       // 10,240,000
  float* T1 = T0 + 10240000;       // 10,240,000
  float* SM = T1 + 10240000;       //  4,000,000  (attn-out, BN stats, packed whh)

  auto gemm = [&](const float* A, const float* B, const float* bias, const float* res,
                  float* C, int M, int N, int K, int act) {
    dim3 g((N + 63) / 64, (M + 63) / 64);
    wmma_gemm_kernel<<<g, 256, 0, stream>>>(A, B, bias, res, C, M, N, K, act);
  };

  // ---- linear0: relu(x@w1) -> relu(@w2) -> @w3 ----
  gemm(x,  l0w1, l0b1, nullptr, T0, 10240, 128, 500, 1);
  gemm(T0, l0w2, l0b2, nullptr, T1, 10240, 256, 128, 1);
  gemm(T1, l0w3, l0b3, nullptr, Z,  10240, 500, 256, 0);

  // ---- conv stack + BN + leaky ----
  { long long tot = 256LL * 64 * 500;
    conv1d_k3<<<(unsigned)((tot + 255) / 256), 256, 0, stream>>>(Z, c11w, c11b, W0, 256, 40, 64, 500, 1); }
  { long long tot = 256LL * 40 * 500;
    conv1d_k3<<<(unsigned)((tot + 255) / 256), 256, 0, stream>>>(W0, c21w, c21b, T0, 256, 64, 40, 500, 0); }
  bn_stats_ch<<<40, 256, 0, stream>>>(T0, SM, SM + 64, 256, 40, 500);
  { long long tot = 256LL * 40 * 500;
    bn_apply_ch<<<(unsigned)((tot + 255) / 256), 256, 0, stream>>>(T0, Z, SM, SM + 64, bn11g, bn11b, 256, 40, 500); }

  // ---- 2 transformer blocks ----
  for (int l = 0; l < 2; ++l) {
    int b = 15 + l * 11;
    const float *ln1g = P(b+0), *ln1b = P(b+1), *qkvw = P(b+2), *outw = P(b+3), *outb = P(b+4);
    const float *ln2g = P(b+5), *ln2b = P(b+6), *ff1w = P(b+7), *ff1b = P(b+8), *ff2w = P(b+9), *ff2b = P(b+10);
    layernorm_kernel<<<10240, 128, 0, stream>>>(Z, ln1g, ln1b, T0, 500);
    gemm(T0, qkvw, nullptr, nullptr, T1, 10240, 192, 500, 0);     // qkv (no bias)
    attn_kernel<<<2048, 256, 0, stream>>>(T1, SM);                // SM = attn out [10240,64]
    gemm(SM, outw, outb, Z, Z, 10240, 500, 64, 0);                // h += o@out_w + b
    layernorm_kernel<<<10240, 128, 0, stream>>>(Z, ln2g, ln2b, T0, 500);
    gemm(T0, ff1w, ff1b, nullptr, T1, 10240, 8, 500, 3);          // gelu
    gemm(T1, ff2w, ff2b, Z, Z, 10240, 500, 8, 0);                 // h += ff
  }

  // ---- 2-layer bidirectional LSTM ----
  size_t lds_bytes = (size_t)(2 * 24576 * 2) + (size_t)(40 * 512 * 4);  // 180224
  __bf16* whp = (__bf16*)SM;   // packed recurrent weights (2MB), SM free here
  for (int l = 0; l < 2; ++l) {
    int b = 37 + l * 6;
    const float *wihf = P(b+0), *whhf = P(b+1), *bf = P(b+2);
    const float *wihb = P(b+3), *whhb = P(b+4), *bb = P(b+5);
    const float* zin = (l == 0) ? Z : T0;
    float* zout = (l == 0) ? T0 : Z;
    int Kin = (l == 0) ? 500 : 1000;
    gemm(zin, wihf, bf, nullptr, W0, 10240, 2000, Kin, 0);        // pre_f
    gemm(zin, wihb, bb, nullptr, W1, 10240, 2000, Kin, 0);        // pre_b
    lstm_pack_whh<<<4096, 256, 0, stream>>>(whhf, whp);
    lstm_dir_kernel<<<1, 512, lds_bytes, stream>>>(W0, whp, zout, 256, 0, 1000, 0);
    lstm_pack_whh<<<4096, 256, 0, stream>>>(whhb, whp);
    lstm_dir_kernel<<<1, 512, lds_bytes, stream>>>(W1, whp, zout, 256, 1, 1000, 500);
  }
  // nodes = Z [10240,1000]

  // ---- GATv2 layer 1 (H=4, C=128) ----
  float* xl1 = W0;
  float* xr1 = W0 + 5242880;
  float* a1 = W1, *w1 = W1 + 700000, *m1 = W1 + 1400000, *dn1 = W1 + 1460000;
  gemm(Z, g1wl, g1bl, nullptr, xl1, 10240, 512, 1000, 0);
  gemm(Z, g1wr, g1br, nullptr, xr1, 10240, 512, 1000, 0);
  fill_kernel<<<(10240 * 4 + 255) / 256, 256, 0, stream>>>(m1, -3.0e38f, 10240 * 4);
  fill_kernel<<<(10240 * 4 + 255) / 256, 256, 0, stream>>>(dn1, 0.f, 10240 * 4);
  fill_kernel<<<(5242880 + 255) / 256, 256, 0, stream>>>(T0, 0.f, 5242880);
  { long long tot = (long long)E * 4;
    gat_edge_score<<<(unsigned)((tot + 255) / 256), 256, 0, stream>>>(xl1, xr1, g1att, src, dst, a1, m1, E, 4, 128);
    gat_edge_exp<<<(unsigned)((tot + 255) / 256), 256, 0, stream>>>(a1, dst, m1, w1, dn1, E, 4); }
  gat_edge_accum<<<E, 256, 0, stream>>>(xl1, w1, dn1, src, dst, T0, E, 4, 128);
  bias_act_kernel<<<(unsigned)((10240LL * 512 + 255) / 256), 256, 0, stream>>>(T0, g1bias, 10240LL * 512, 512, 1);
  // nodes1 = T0 [10240,512]

  // ---- GATv2 layer 2 (H=4, C=250) ----
  float* xl2 = W0;
  float* xr2 = W0 + 10240000;
  float* a2 = W1, *w2 = W1 + 700000, *m2 = W1 + 1400000, *dn2 = W1 + 1460000;
  gemm(T0, g2wl, g2bl, nullptr, xl2, 10240, 1000, 512, 0);
  gemm(T0, g2wr, g2br, nullptr, xr2, 10240, 1000, 512, 0);
  fill_kernel<<<(10240 * 4 + 255) / 256, 256, 0, stream>>>(m2, -3.0e38f, 10240 * 4);
  fill_kernel<<<(10240 * 4 + 255) / 256, 256, 0, stream>>>(dn2, 0.f, 10240 * 4);
  fill_kernel<<<(10240000 + 255) / 256, 256, 0, stream>>>(Z, 0.f, 10240000);
  { long long tot = (long long)E * 4;
    gat_edge_score<<<(unsigned)((tot + 255) / 256), 256, 0, stream>>>(xl2, xr2, g2att, src, dst, a2, m2, E, 4, 250);
    gat_edge_exp<<<(unsigned)((tot + 255) / 256), 256, 0, stream>>>(a2, dst, m2, w2, dn2, E, 4); }
  gat_edge_accum<<<E, 256, 0, stream>>>(xl2, w2, dn2, src, dst, Z, E, 4, 250);
  bias_act_kernel<<<(unsigned)((10240LL * 1000 + 255) / 256), 256, 0, stream>>>(Z, g2bias, 10240LL * 1000, 1000, 0);

  // ---- BN over nodes dim + relu, pool per graph, L2-normalize ----
  bn_stats_col<<<1000, 256, 0, stream>>>(Z, SM, SM + 1024, 10240, 1000);
  bn_apply_col<<<(unsigned)((10240LL * 1000 + 255) / 256), 256, 0, stream>>>(Z, SM, SM + 1024, bn2g, bn2b, 10240, 1000);
  pool_kernel<<<(256 * 1000 + 255) / 256, 256, 0, stream>>>(Z, (float*)d_out, 256, 40, 1000);
  rownorm_kernel<<<256, 256, 0, stream>>>((float*)d_out, 1000);
}
